// CrossAttenHead_72198400245928
// MI455X (gfx1250) — compile-verified, matrix operand order checked by reading, not run on hardware
//
#include <hip/hip_runtime.h>
#include <hip/hip_bf16.h>

// ---------------------------------------------------------------------------
// CrossAttenHead forward for MI455X (gfx1250, wave32, WMMA).
// All GEMMs run on v_wmma_f32_16x16x32_f16 (f16 in, f32 accumulate).
// B=8, N=100, C=256, HW=16384, heads=8, dh=32, FFN=2048, NCLS=134, NK=103.
// Big GEMMs (mask pooling, mask projection) use 4x2 register-blocked tiles
// per wave to cut L2 panel re-reads ~4x; small weight GEMMs use 1x1 tiles.
// All K-loops are branch-free (K padded to multiples of 32 where needed).
// ---------------------------------------------------------------------------

typedef __attribute__((ext_vector_type(16))) _Float16 v16h;
typedef __attribute__((ext_vector_type(8)))  float    v8f;

namespace cfg {
constexpr int B    = 8;
constexpr int N    = 100;
constexpr int C    = 256;
constexpr int HW   = 128 * 128;
constexpr int NK   = 103;    // N + K (dysep dynamic weights)
constexpr int KP   = 128;    // padded K for the pointwise-mix GEMM
constexpr int FFN  = 2048;
constexpr int NCLS = 134;    // NUM_CLASSES + 1
}

enum { GF_RELU = 2 };

// Fragment layouts per ISA 7.12.2 (16-bit A 16x32): lane L holds row M=L&15;
// half e holds K = (e&8?16:0) + 8*(L>>4) + (e&7) -> two contiguous 8-float
// runs per lane -> 4x float4 (global_load_b128) per matrix per 32-K step.
// B assumed transpose-symmetric (lane = column N, same K packing).
// C/D: float r of v8f -> M = r + 8*(L>>4), N = L&15.
// Out-of-range M/N indices are *clamped* on load (a clamped A-row/B-column
// only pollutes D rows/cols the epilogue never stores) -> branch-free K-loop.

// ---------------------------------------------------------------------------
// 1x1-tile GEMM for the small GEMMs.  D[b,m,n] = sum_k A*B (+bias, +relu)
//   BUNITK=1 : B element (k,n) at Bb[n*b_ns + k];  =0 : at Bb[k*b_ks + n]
// K must be a multiple of 32; A row stride must be a multiple of 4 floats.
// ---------------------------------------------------------------------------
template<bool BUNITK>
__global__ __launch_bounds__(128)
void wmma_gemm_kernel(const float* __restrict__ A, int a_bs, int a_ms,
                      const float* __restrict__ Bm, int b_bs, int b_ks, int b_ns,
                      float* __restrict__ D, int d_bs, int d_ms,
                      const float* __restrict__ bias,
                      int M, int Nn, int K,
                      int mtiles, int ntiles, int total_tiles, int flags)
{
    int tile = blockIdx.x * 4 + (threadIdx.x >> 5);
    if (tile >= total_tiles) return;              // wave-uniform exit
    const int lane = threadIdx.x & 31;

    int nt = tile % ntiles; tile /= ntiles;
    int mt = tile % mtiles;
    int b  = tile / mtiles;

    int am = mt * 16 + (lane & 15); if (am >= M)  am = M - 1;
    int bn = nt * 16 + (lane & 15); if (bn >= Nn) bn = Nn - 1;

    const float* pA = A + (long)b * a_bs + (long)am * a_ms;
    const float* pB = BUNITK ? (Bm + (long)b * b_bs + (long)bn * b_ns)
                             : (Bm + (long)b * b_bs + bn);
    const int khalf = (lane >> 4) << 3;

    v8f acc = {};
#pragma unroll 2
    for (int k0 = 0; k0 < K; k0 += 32) {
        float abuf[16], bbuf[16];
        const float* qa = pA + k0 + khalf;
        *(float4*)(abuf + 0)  = *(const float4*)(qa + 0);
        *(float4*)(abuf + 4)  = *(const float4*)(qa + 4);
        *(float4*)(abuf + 8)  = *(const float4*)(qa + 16);
        *(float4*)(abuf + 12) = *(const float4*)(qa + 20);
        if (BUNITK) {
            const float* qb = pB + k0 + khalf;
            *(float4*)(bbuf + 0)  = *(const float4*)(qb + 0);
            *(float4*)(bbuf + 4)  = *(const float4*)(qb + 4);
            *(float4*)(bbuf + 8)  = *(const float4*)(qb + 16);
            *(float4*)(bbuf + 12) = *(const float4*)(qb + 20);
        } else {
#pragma unroll
            for (int e = 0; e < 16; ++e) {
                int kk = k0 + ((e & 8) ? 16 : 0) + khalf + (e & 7);
                bbuf[e] = pB[kk * b_ks];
            }
        }
        v16h af, bf;
#pragma unroll
        for (int e = 0; e < 16; ++e) {
            af[e] = (_Float16)abuf[e];
            bf[e] = (_Float16)bbuf[e];
        }
        acc = __builtin_amdgcn_wmma_f32_16x16x32_f16(
                  false, af, false, bf, (short)0, acc, false, false);
    }

    const int n = nt * 16 + (lane & 15);
    const bool n_ok = n < Nn;
    const int mrow0 = mt * 16 + ((lane >> 4) << 3);
    const int mleft = M - mrow0;
    float* pD = D + (long)b * d_bs + (long)mrow0 * d_ms + n;
#pragma unroll
    for (int r = 0; r < 8; ++r) {
        if (n_ok && r < mleft) {
            float v = acc[r];
            if (bias) v += bias[n];
            if (flags & GF_RELU) v = fmaxf(v, 0.0f);
            pD[r * d_ms] = v;
        }
    }
}

// ---------------------------------------------------------------------------
// MT x NT register-blocked GEMM for the two feature-sized GEMMs.
// Each wave owns an MT*16 x NT*16 output block: A fragments are reused NT
// times, B fragments MT times -> panel L2 traffic drops by those factors.
//   BIN    : binarize A on load (sigmoid(x)>0.5 <=> x>0)   [mask pooling]
//   ATOMIC : K-split reduction via global_atomic_add_f32 (D pre-zeroed)
// K must be a multiple of 32 (holds: HW=16384, C=256). No bias/relu needed.
// ---------------------------------------------------------------------------
template<int MT, int NT, bool BUNITK, bool BIN, bool ATOMIC>
__global__ __launch_bounds__(128)
void wmma_gemm_blk(const float* __restrict__ A, int a_bs, int a_ms,
                   const float* __restrict__ Bm, int b_bs, int b_ks, int b_ns,
                   float* __restrict__ D, int d_bs, int d_ms,
                   int M, int Nn, int K,
                   int mgroups, int ngroups, int ksplit, int total_tiles)
{
    int tile = blockIdx.x * 4 + (threadIdx.x >> 5);
    if (tile >= total_tiles) return;              // wave-uniform exit
    const int lane = threadIdx.x & 31;

    int ks = tile % ksplit;  tile /= ksplit;
    int ng = tile % ngroups; tile /= ngroups;
    int mg = tile % mgroups;
    int b  = tile / mgroups;

    int kchunk = (ksplit > 1) ? (((K + ksplit - 1) / ksplit + 31) & ~31) : K;
    int kbeg = ks * kchunk;
    int kend = kbeg + kchunk; if (kend > K) kend = K;

    const int khalf = (lane >> 4) << 3;

    const float* pA[MT];
#pragma unroll
    for (int i = 0; i < MT; ++i) {
        int am = (mg * MT + i) * 16 + (lane & 15);
        if (am >= M) am = M - 1;
        pA[i] = A + (long)b * a_bs + (long)am * a_ms;
    }
    const float* pB[NT];
#pragma unroll
    for (int j = 0; j < NT; ++j) {
        int bn = (ng * NT + j) * 16 + (lane & 15);
        if (bn >= Nn) bn = Nn - 1;
        pB[j] = BUNITK ? (Bm + (long)b * b_bs + (long)bn * b_ns)
                       : (Bm + (long)b * b_bs + bn);
    }

    v8f zero = {};
    v8f acc[MT][NT];
#pragma unroll
    for (int i = 0; i < MT; ++i)
#pragma unroll
        for (int j = 0; j < NT; ++j) acc[i][j] = zero;

    for (int k0 = kbeg; k0 < kend; k0 += 32) {
        v16h af[MT], bf[NT];
#pragma unroll
        for (int i = 0; i < MT; ++i) {
            float buf[16];
            const float* qa = pA[i] + k0 + khalf;
            *(float4*)(buf + 0)  = *(const float4*)(qa + 0);
            *(float4*)(buf + 4)  = *(const float4*)(qa + 4);
            *(float4*)(buf + 8)  = *(const float4*)(qa + 16);
            *(float4*)(buf + 12) = *(const float4*)(qa + 20);
#pragma unroll
            for (int e = 0; e < 16; ++e) {
                float v = buf[e];
                if (BIN) v = (v > 0.0f) ? 1.0f : 0.0f;
                af[i][e] = (_Float16)v;
            }
        }
#pragma unroll
        for (int j = 0; j < NT; ++j) {
            float buf[16];
            if (BUNITK) {
                const float* qb = pB[j] + k0 + khalf;
                *(float4*)(buf + 0)  = *(const float4*)(qb + 0);
                *(float4*)(buf + 4)  = *(const float4*)(qb + 4);
                *(float4*)(buf + 8)  = *(const float4*)(qb + 16);
                *(float4*)(buf + 12) = *(const float4*)(qb + 20);
            } else {
#pragma unroll
                for (int e = 0; e < 16; ++e) {
                    int kk = k0 + ((e & 8) ? 16 : 0) + khalf + (e & 7);
                    buf[e] = pB[j][kk * b_ks];
                }
            }
#pragma unroll
            for (int e = 0; e < 16; ++e) bf[j][e] = (_Float16)buf[e];
        }
#pragma unroll
        for (int i = 0; i < MT; ++i)
#pragma unroll
            for (int j = 0; j < NT; ++j)
                acc[i][j] = __builtin_amdgcn_wmma_f32_16x16x32_f16(
                    false, af[i], false, bf[j], (short)0, acc[i][j], false, false);
    }

#pragma unroll
    for (int j = 0; j < NT; ++j) {
        int n = (ng * NT + j) * 16 + (lane & 15);
        bool n_ok = n < Nn;
#pragma unroll
        for (int i = 0; i < MT; ++i) {
            int mrow0 = (mg * MT + i) * 16 + ((lane >> 4) << 3);
            int mleft = M - mrow0;
            float* pD = D + (long)b * d_bs + (long)mrow0 * d_ms + n;
#pragma unroll
            for (int r = 0; r < 8; ++r) {
                if (n_ok && r < mleft) {
                    if (ATOMIC) atomicAdd(pD + r * d_ms, acc[i][j][r]);
                    else        pD[r * d_ms] = acc[i][j][r];
                }
            }
        }
    }
}

// Copy pointwise weights dyw[:,3:103] into aligned padded buffer [rows,128],
// zero-filling columns 100..127 (padded-K trick -> branch-free pw GEMM).
__global__ void split_pw_kernel(const float* __restrict__ dyw,
                                float* __restrict__ pwA, int total)
{
    int i = blockIdx.x * blockDim.x + threadIdx.x;
    if (i >= total) return;
    int c = i & (cfg::KP - 1);
    int r = i >> 7;
    pwA[i] = (c < cfg::N) ? dyw[(long)r * cfg::NK + 3 + c] : 0.0f;
}

// LayerNorm over last dim (C=256), optional residual input and trailing ReLU.
__global__ __launch_bounds__(256)
void ln_kernel(const float* __restrict__ x, const float* __restrict__ res,
               const float* __restrict__ g, const float* __restrict__ bb,
               float* __restrict__ out, int relu)
{
    __shared__ float red[256];
    int row = blockIdx.x, tid = threadIdx.x;
    long idx = (long)row * 256 + tid;
    float v = x[idx] + (res ? res[idx] : 0.0f);
    red[tid] = v; __syncthreads();
    for (int off = 128; off > 0; off >>= 1) {
        if (tid < off) red[tid] += red[tid + off];
        __syncthreads();
    }
    float mu = red[0] * (1.0f / 256.0f); __syncthreads();
    float d = v - mu;
    red[tid] = d * d; __syncthreads();
    for (int off = 128; off > 0; off >>= 1) {
        if (tid < off) red[tid] += red[tid + off];
        __syncthreads();
    }
    float var = red[0] * (1.0f / 256.0f);
    float y = d * rsqrtf(var + 1e-5f) * g[tid] + bb[tid];
    if (relu) y = fmaxf(y, 0.0f);
    out[idx] = y;
}

// Dynamic depthwise conv over channel axis (K=3, pad 1) + ReLU.
__global__ void dconv_kernel(const float* __restrict__ val,
                             const float* __restrict__ dyw,
                             float* __restrict__ out, int total)
{
    int i = blockIdx.x * blockDim.x + threadIdx.x;
    if (i >= total) return;
    int c = i & 255;
    int r = i >> 8;
    const float* w = dyw + (long)r * cfg::NK;
    const float* v = val + ((long)r << 8);
    float vm = (c > 0)   ? v[c - 1] : 0.0f;
    float v0 = v[c];
    float vp = (c < 255) ? v[c + 1] : 0.0f;
    out[i] = fmaxf(w[0] * vm + w[1] * v0 + w[2] * vp, 0.0f);
}

// Per-(b,h,q) softmax attention, seq N=100, dh=32, qkv packed [B,N,768].
__global__ __launch_bounds__(128)
void attn_kernel(const float* __restrict__ qkv, float* __restrict__ outp)
{
    __shared__ float qv[32];
    __shared__ float sc[128];
    __shared__ float red[128];
    int q = blockIdx.x % cfg::N;
    int h = (blockIdx.x / cfg::N) % 8;
    int b = blockIdx.x / (cfg::N * 8);
    int tid = threadIdx.x;
    long base = (long)b * cfg::N * 768;
    if (tid < 32) qv[tid] = qkv[base + (long)q * 768 + h * 32 + tid];
    __syncthreads();

    float s = -3.0e38f;
    if (tid < cfg::N) {
        const float* kp = qkv + base + (long)tid * 768 + 256 + h * 32;
        float dot = 0.0f;
#pragma unroll
        for (int d = 0; d < 32; ++d) dot += qv[d] * kp[d];
        s = dot * 0.17677669529663687f;   // 1/sqrt(32)
    }
    red[tid] = s; __syncthreads();
    for (int off = 64; off > 0; off >>= 1) {
        if (tid < off) red[tid] = fmaxf(red[tid], red[tid + off]);
        __syncthreads();
    }
    float mx = red[0]; __syncthreads();
    float e = (tid < cfg::N) ? __expf(s - mx) : 0.0f;
    sc[tid] = e; red[tid] = e; __syncthreads();
    for (int off = 64; off > 0; off >>= 1) {
        if (tid < off) red[tid] += red[tid + off];
        __syncthreads();
    }
    float inv = 1.0f / red[0];
    if (tid < 32) {
        const float* vp = qkv + base + 512 + h * 32 + tid;
        float o = 0.0f;
        for (int j = 0; j < cfg::N; ++j) o += sc[j] * vp[(long)j * 768];
        outp[((long)b * cfg::N + q) * 256 + h * 32 + tid] = o * inv;
    }
}

// ---------------------------------------------------------------------------
// 1x1-tile launcher.  bstrided=true -> B has unit-N stride (strided K reads)
static inline void gemm1(hipStream_t st, bool bstrided,
                         const float* A, int abs_, int ams,
                         const float* Bm, int bbs, int bks, int bns,
                         float* D, int dbs, int dms,
                         const float* bias, int M, int Nn, int K,
                         int batches, int flags)
{
    int mt = (M + 15) / 16, nt = (Nn + 15) / 16;
    int total = batches * mt * nt;
    dim3 grid((unsigned)((total + 3) / 4)), blk(128);
    if (bstrided)
        wmma_gemm_kernel<false><<<grid, blk, 0, st>>>(A, abs_, ams,
            Bm, bbs, bks, bns, D, dbs, dms, bias, M, Nn, K, mt, nt, total, flags);
    else
        wmma_gemm_kernel<true ><<<grid, blk, 0, st>>>(A, abs_, ams,
            Bm, bbs, bks, bns, D, dbs, dms, bias, M, Nn, K, mt, nt, total, flags);
}

extern "C" void kernel_launch(void* const* d_in, const int* in_sizes, int n_in,
                              void* d_out, int out_size, void* d_ws, size_t ws_size,
                              hipStream_t stream)
{
    using namespace cfg;
    (void)out_size; (void)ws_size;

    // ---- map inputs (hedge against two plausible pytree flattenings) ----
    const float* features = (const float*)d_in[0];           // [B,C,H,W]
    const float* prop;                                       // [B,N,C]
    const float* maskp;                                      // [B,N,H,W]
    int pbase;
    if (in_sizes[1] == B * N * C) {            // insertion order: f, prop, mask, params...
        prop  = (const float*)d_in[1];
        maskp = (const float*)d_in[2];
        pbase = 3;
    } else {                                   // fully sorted: f, mask, params..., prop
        maskp = (const float*)d_in[1];
        prop  = (const float*)d_in[n_in - 1];
        pbase = 2;
    }
    const float* P[34];
    for (int i = 0; i < 34; ++i) P[i] = (const float*)d_in[pbase + i];
    // alphabetical pytree order of the params dict:
    const float *cls_ln_b = P[0],  *cls_ln_g = P[1],  *cls_fc_w = P[2];
    const float *fA_b = P[3], *fA_lnb = P[4], *fA_lng = P[5], *fA_w = P[6];
    const float *f_nb = P[7], *f_ng = P[8];
    const float *fc_cls_b = P[9],  *fc_cls_w = P[10];
    const float *fc_mask_b = P[11], *fc_mask_w = P[12];
    const float *ffn_b1 = P[13], *ffn_b2 = P[14], *ffn_nb = P[15], *ffn_ng = P[16];
    const float *ffn_w1 = P[17], *ffn_w2 = P[18];
    const float *kA_b = P[19], *kA_lnb = P[20], *kA_lng = P[21], *kA_w = P[22];
    const float *k_nb = P[23], *k_ng = P[24];
    const float *m_ln_b = P[25], *m_ln_g = P[26], *mask_fc_w = P[27];
    const float *s_inb = P[28], *s_inw = P[29], *s_outb = P[30], *s_outw = P[31];
    const float *s_nb = P[32], *s_ng = P[33];

    // ---- workspace carve-out (all offsets multiples of 4 floats = 16B) ----
    float* w = (float*)d_ws;
    auto alloc = [&](long nfl) { float* p = w; w += nfl; return p; };
    const long BN = (long)B * N;
    float* f0    = alloc(BN * C);
    float* dywF  = alloc(BN * NK + 4);
    float* convF = alloc(BN * C);
    float* pwF   = alloc(BN * C);
    float* t1    = alloc(BN * C);
    float* f1    = alloc(BN * C);
    float* dywK  = alloc(BN * NK + 4);
    float* convK = alloc(BN * C);
    float* pwK   = alloc(BN * C);
    float* t2    = alloc(BN * C);
    float* k2    = alloc(BN * C);
    float* pwA   = alloc(BN * KP);       // padded pointwise weights
    float* qkv   = alloc(BN * 3 * C);
    float* mhaO  = alloc(BN * C);
    float* attn  = alloc(BN * C);
    float* k3    = alloc(BN * C);
    float* ffnH  = alloc(BN * FFN);
    float* ffnO  = alloc(BN * C);
    float* clsA  = alloc(BN * C);
    float* clsB  = alloc(BN * C);
    float* mskA  = alloc(BN * C);
    float* mskB  = alloc(BN * C);
    float* mkern = alloc(BN * C);

    float* out      = (float*)d_out;
    float* out_cls  = out;                          // [B,N,134]
    float* out_mask = out + (long)B * N * NCLS;     // [B,N,HW]
    float* out_obj  = out_mask + (long)B * N * HW;  // [B,N,C]

    const int rows = (int)BN;
    const int dctotal = (int)(BN * C);
    const int pwtotal = (int)(BN * KP);
    const int mtiles = (N + 15) / 16;               // 7

    // 1) f0[b,n,c] = sum_hw [maskp>0] * features
    //    4x2 blocked, K-split 32, atomic reduce (f0 pre-zeroed)
    hipMemsetAsync(f0, 0, BN * C * sizeof(float), stream);
    {
        const int MT = 4, NT = 2, KSPLIT = 32;
        int mg = (mtiles + MT - 1) / MT;            // 2
        int ng = (C / 16 + NT - 1) / NT;            // 8
        int total = B * mg * ng * KSPLIT;
        wmma_gemm_blk<MT, NT, true, true, true>
            <<<dim3((unsigned)((total + 3) / 4)), dim3(128), 0, stream>>>(
                maskp, N * HW, HW,
                features, C * HW, 1, HW,
                f0, N * C, C,
                N, C, HW, mg, ng, KSPLIT, total);
    }

    // 2) dysep #1 (query = proposal kernels, value = f0)
    //    pw GEMM: A = padded pw weights [rows,128], zero cols 100..127 kill
    //    the garbage B rows read for k in [100,128) -> branch-free K-loop.
    gemm1(stream, false, prop, N * C, C,  fA_w, 0, 1, C,
          dywF, N * NK, NK, fA_b, N, NK, C, B, 0);
    dconv_kernel<<<(dctotal + 255) / 256, 256, 0, stream>>>(f0, dywF, convF, dctotal);
    split_pw_kernel<<<(pwtotal + 255) / 256, 256, 0, stream>>>(dywF, pwA, pwtotal);
    gemm1(stream, true, pwA, N * KP, KP,
          convF, N * C, C, 1,
          pwF, N * C, C, nullptr, N, C, KP, B, 0);
    ln_kernel<<<rows, 256, 0, stream>>>(pwF, nullptr, fA_lng, fA_lnb, t1, 0);
    ln_kernel<<<rows, 256, 0, stream>>>(f0, t1, f_ng, f_nb, f1, 0);    // f1 = LN(f0+t1)

    // 3) dysep #2 (value = f1) ; k2 = LN(f1 + t2)
    gemm1(stream, false, prop, N * C, C,  kA_w, 0, 1, C,
          dywK, N * NK, NK, kA_b, N, NK, C, B, 0);
    dconv_kernel<<<(dctotal + 255) / 256, 256, 0, stream>>>(f1, dywK, convK, dctotal);
    split_pw_kernel<<<(pwtotal + 255) / 256, 256, 0, stream>>>(dywK, pwA, pwtotal);
    gemm1(stream, true, pwA, N * KP, KP,
          convK, N * C, C, 1,
          pwK, N * C, C, nullptr, N, C, KP, B, 0);
    ln_kernel<<<rows, 256, 0, stream>>>(pwK, nullptr, kA_lng, kA_lnb, t2, 0);
    ln_kernel<<<rows, 256, 0, stream>>>(f1, t2, k_ng, k_nb, k2, 0);

    // 4) self-attention: qkv proj -> softmax attention -> out proj -> LN
    gemm1(stream, false, k2, N * C, C,  s_inw, 0, 1, C,
          qkv, N * 3 * C, 3 * C, s_inb, N, 3 * C, C, B, 0);
    attn_kernel<<<B * 8 * N, 128, 0, stream>>>(qkv, mhaO);
    gemm1(stream, false, mhaO, N * C, C,  s_outw, 0, 1, C,
          attn, N * C, C, s_outb, N, C, C, B, 0);
    ln_kernel<<<rows, 256, 0, stream>>>(k2, attn, s_ng, s_nb, k3, 0);

    // 5) FFN + LN -> obj_feat (written straight into d_out)
    gemm1(stream, false, k3, N * C, C,  ffn_w1, 0, 1, C,
          ffnH, N * FFN, FFN, ffn_b1, N, FFN, C, B, GF_RELU);
    gemm1(stream, false, ffnH, N * FFN, FFN,  ffn_w2, 0, 1, FFN,
          ffnO, N * C, C, ffn_b2, N, C, FFN, B, 0);
    ln_kernel<<<rows, 256, 0, stream>>>(k3, ffnO, ffn_ng, ffn_nb, out_obj, 0);

    // 6) classification branch
    gemm1(stream, false, out_obj, N * C, C,  cls_fc_w, 0, 1, C,
          clsA, N * C, C, nullptr, N, C, C, B, 0);
    ln_kernel<<<rows, 256, 0, stream>>>(clsA, nullptr, cls_ln_g, cls_ln_b, clsB, 1);
    gemm1(stream, false, clsB, N * C, C,  fc_cls_w, 0, 1, C,
          out_cls, N * NCLS, NCLS, fc_cls_b, N, NCLS, C, B, 0);

    // 7) mask branch -> kernels -> mask prediction over all pixels
    gemm1(stream, false, out_obj, N * C, C,  mask_fc_w, 0, 1, C,
          mskA, N * C, C, nullptr, N, C, C, B, 0);
    ln_kernel<<<rows, 256, 0, stream>>>(mskA, nullptr, m_ln_g, m_ln_b, mskB, 1);
    gemm1(stream, false, mskB, N * C, C,  fc_mask_w, 0, 1, C,
          mkern, N * C, C, fc_mask_b, N, C, C, B, 0);

    // new_mask_preds[b,q,hw] = sum_c mkern[b,q,c] * features[b,c,hw]
    // 4x2 blocked: features panel re-read drops from 7x to 2x
    {
        const int MT = 4, NT = 2;
        int mg = (mtiles + MT - 1) / MT;            // 2
        int ng = (HW / 16 + NT - 1) / NT;           // 512
        int total = B * mg * ng;
        wmma_gemm_blk<MT, NT, false, false, false>
            <<<dim3((unsigned)((total + 3) / 4)), dim3(128), 0, stream>>>(
                mkern, N * C, C,
                features, C * HW, HW, 1,
                out_mask, N * HW, HW,
                N, HW, C, mg, ng, 1, total);
    }
}
